// AFTFull_41085657153699
// MI455X (gfx1250) — matrix operand order repaired and checked by reading, not compiled, not run on hardware
//
#include <hip/hip_runtime.h>
#include <hip/hip_bf16.h>
#include <math.h>

// AFT-Full: B=32, T=1024, D=512, fp32 in/out.
// Memory-bound (~206 GFLOP vs a few hundred MB @ 23.3 TB/s) ->
//  * one-time f32->f16 conversion of x and weights (halves GEMM read traffic)
//  * f16 WMMA (fp32 accumulate) everywhere
//  * CDNA5 async global->LDS copies (global_load_async_to_lds_b64, ASYNCcnt)
//  * double-buffered LDS + partial s_wait_asynccnt N waits so the async DMA
//    for tile k+1 overlaps the WMMA work on tile k.

typedef __attribute__((ext_vector_type(16))) _Float16 v16h;
typedef __attribute__((ext_vector_type(8)))  float    v8f;

#define Bdim 32
#define Tdim 1024
#define Ddim 512
#define TD   (Tdim * Ddim)          // 524288
#define BTD  (Bdim * Tdim * Ddim)   // 16777216

// ---- CDNA5 async global->LDS copy (CDNA5 ISA) -------------
__device__ __forceinline__ void async_b64(unsigned lds_off, const void* g) {
  asm volatile("global_load_async_to_lds_b64 %0, %1, off"
               :: "v"(lds_off), "v"((unsigned long long)(size_t)g)
               : "memory");
}
__device__ __forceinline__ void wait_async0() {   // all asyncs done
  asm volatile("s_wait_asynccnt 0" ::: "memory");
}
__device__ __forceinline__ void wait_async3() {   // older group of 3 done
  asm volatile("s_wait_asynccnt 3" ::: "memory");
}
__device__ __forceinline__ void wait_async5() {   // older group of 5 done
  asm volatile("s_wait_asynccnt 5" ::: "memory");
}

// ---------------------------------------------------------------------------
// f32 -> f16 pairwise conversion (x and the four weight matrices)
// ---------------------------------------------------------------------------
__global__ __launch_bounds__(256) void cvt_kernel(const float* __restrict__ in,
                                                  _Float16* __restrict__ out,
                                                  int npairs) {
  int i = blockIdx.x * 256 + threadIdx.x;
  if (i < npairs) {
    float2 v = ((const float2*)in)[i];
    union { _Float16 h[2]; unsigned u; } pk;
    pk.h[0] = (_Float16)v.x; pk.h[1] = (_Float16)v.y;
    ((unsigned*)out)[i] = pk.u;
  }
}

// ---------------------------------------------------------------------------
// C[m,e] = sum_d A[m,d] * W[e,d] + bias[e]   (A, W already f16)
// Block = 256 threads = 8 waves (wave32), tile 32(M) x 64(N), K step 32.
// Double-buffered async tile staging (3 async b64 per thread per tile).
// ---------------------------------------------------------------------------
__global__ __launch_bounds__(256) void gemm_h(const _Float16* __restrict__ A,
                                              const _Float16* __restrict__ W,
                                              const float* __restrict__ bias,
                                              float* __restrict__ C) {
  __shared__ _Float16 sA[2][32 * 36];   // [m 32][k 32], row stride 36 halves
  __shared__ _Float16 sB[2][64 * 36];   // [e 64][k 32]

  const int tid  = threadIdx.x;
  const int lane = tid & 31;
  const int wave = tid >> 5;
  const int wm   = wave & 1;
  const int wn   = wave >> 1;
  const int m0   = blockIdx.x * 32;
  const int e0   = blockIdx.y * 64;

  // async copy plan: 8B chunks. sA: 32 rows * 8 chunks = 256 -> 1/thread.
  const int arow = tid >> 3, ach = tid & 7;
  const _Float16* agp = A + (size_t)(m0 + arow) * Ddim + ach * 4;
  // sB: 64 rows * 8 chunks = 512 -> 2/thread.
  const int br0 = tid >> 3,         bc0 = tid & 7;
  const int br1 = (tid + 256) >> 3, bc1 = tid & 7;
  const _Float16* bgp0 = W + (size_t)(e0 + br0) * Ddim + bc0 * 4;
  const _Float16* bgp1 = W + (size_t)(e0 + br1) * Ddim + bc1 * 4;

  unsigned aoff[2], boff0[2], boff1[2];
  #pragma unroll
  for (int bb = 0; bb < 2; ++bb) {
    aoff[bb]  = (unsigned)(size_t)&sA[bb][arow * 36 + ach * 4];
    boff0[bb] = (unsigned)(size_t)&sB[bb][br0 * 36 + bc0 * 4];
    boff1[bb] = (unsigned)(size_t)&sB[bb][br1 * 36 + bc1 * 4];
  }

  // ISA 7.12.2 fragment mapping (wave32)
  const int mfrag = lane & 15;
  const int koff  = (lane >> 4) * 8;
  const int kb    = (lane >> 4) * 16;

  v8f acc = {};

  // prologue: prefetch tile 0 into buffer 0
  async_b64(aoff[0],  agp);
  async_b64(boff0[0], bgp0);
  async_b64(boff1[0], bgp1);

  #pragma unroll 2
  for (int it = 0; it < Ddim / 32; ++it) {
    const int cur = it & 1;
    const int k0  = it * 32;
    if (k0 + 32 < Ddim) {          // prefetch next tile into other buffer
      const int nxt = cur ^ 1;
      async_b64(aoff[nxt],  agp  + k0 + 32);
      async_b64(boff0[nxt], bgp0 + k0 + 32);
      async_b64(boff1[nxt], bgp1 + k0 + 32);
      wait_async3();               // only tile `it`'s group must have landed
    } else {
      wait_async0();
    }
    __syncthreads();

    v16h af, bf;
    const _Float16* ar = &sA[cur][(wm * 16 + mfrag) * 36];
    #pragma unroll
    for (int j = 0; j < 8; ++j) {
      af[j]     = ar[koff + j];
      af[8 + j] = ar[16 + koff + j];
    }
    const _Float16* br = &sB[cur][(wn * 16 + (lane & 15)) * 36];
    #pragma unroll
    for (int j = 0; j < 8; ++j) { bf[j] = br[koff + j]; bf[8 + j] = br[16 + koff + j]; }

    acc = __builtin_amdgcn_wmma_f32_16x16x32_f16(false, af, false, bf,
                                                 (short)0, acc, false, false);
    __syncthreads();               // all reads of buffer `cur` done
  }

  const int cn   = e0 + wn * 16 + (lane & 15);
  const float bv = bias[cn];
  const int crow = m0 + wm * 16 + (lane >> 4) * 8;
  #pragma unroll
  for (int r = 0; r < 8; ++r)
    C[(size_t)(crow + r) * Ddim + cn] = acc[r] + bv;
}

// ---------------------------------------------------------------------------
// maxK[t,d] = max_b K[b,t,d]
// ---------------------------------------------------------------------------
__global__ __launch_bounds__(256) void maxk_kernel(const float* __restrict__ Kb,
                                                   float* __restrict__ maxK) {
  unsigned i = blockIdx.x * 256 + threadIdx.x;
  float m = Kb[i];
  #pragma unroll 4
  for (int b = 1; b < Bdim; ++b)
    m = fmaxf(m, Kb[(size_t)b * TD + i]);
  maxK[i] = m;
}

// ---------------------------------------------------------------------------
// exp_w[t,s] = exp(w[t,s] - rowmax) -> f16
// ---------------------------------------------------------------------------
__global__ __launch_bounds__(256) void expw_kernel(const float* __restrict__ w,
                                                   _Float16* __restrict__ expw) {
  __shared__ float red[8];
  const int t = blockIdx.x;
  const float* row = w + (size_t)t * Tdim;
  float m = -INFINITY;
  for (int s = threadIdx.x; s < Tdim; s += 256) m = fmaxf(m, row[s]);
  #pragma unroll
  for (int off = 16; off > 0; off >>= 1)
    m = fmaxf(m, __shfl_down(m, off, 32));
  if ((threadIdx.x & 31) == 0) red[threadIdx.x >> 5] = m;
  __syncthreads();
  m = red[0];
  #pragma unroll
  for (int j = 1; j < 8; ++j) m = fmaxf(m, red[j]);
  for (int s = threadIdx.x; s < Tdim; s += 256)
    expw[(size_t)t * Tdim + s] = (_Float16)__expf(row[s] - m);
}

// ---------------------------------------------------------------------------
// EK = exp(K - maxK) (f16), EKV = EK * V (f16)
// ---------------------------------------------------------------------------
__global__ __launch_bounds__(256) void ekv_kernel(const float* __restrict__ K,
                                                  const float* __restrict__ V,
                                                  const float* __restrict__ maxK,
                                                  _Float16* __restrict__ EK,
                                                  _Float16* __restrict__ EKV) {
  size_t i = (size_t)blockIdx.x * 256 + threadIdx.x;
  size_t td = i & (size_t)(TD - 1);
  float ek = __expf(K[i] - maxK[td]);
  EK[i]  = (_Float16)ek;
  EKV[i] = (_Float16)(ek * V[i]);
}

// ---------------------------------------------------------------------------
// num = expw @ EKV, den = expw @ EK (shared A-fragment, two WMMA accums),
// Yt = sigmoid(Q) * num/den  -> f16.  Grid: (T/32, D/64, B).
// Double-buffered async tile staging (5 async b64 per thread per tile).
// ---------------------------------------------------------------------------
__global__ __launch_bounds__(256) void aft_attn(const _Float16* __restrict__ expw,
                                                const _Float16* __restrict__ EK,
                                                const _Float16* __restrict__ EKV,
                                                const float* __restrict__ Q,
                                                _Float16* __restrict__ Yt) {
  __shared__ _Float16 sA [2][32 * 36];   // expw tile [t 32][s 32]
  __shared__ _Float16 sEK[2][32 * 68];   // [s 32][d 64], row stride 68
  __shared__ _Float16 sKV[2][32 * 68];

  const int tid  = threadIdx.x;
  const int lane = tid & 31;
  const int wave = tid >> 5;
  const int wm   = wave & 1;
  const int wn   = wave >> 1;
  const int t0   = blockIdx.x * 32;
  const int d0   = blockIdx.y * 64;
  const int b    = blockIdx.z;

  const _Float16* EKb  = EK  + (size_t)b * TD;
  const _Float16* EKVb = EKV + (size_t)b * TD;

  // async copy plan
  const int arow = tid >> 3, ach = tid & 7;          // sA: 256 chunks of 8B
  const _Float16* agp = expw + (size_t)(t0 + arow) * Tdim + ach * 4;
  const int er0 = tid >> 4,         ec0 = tid & 15;  // sEK/sKV: 512 chunks
  const int er1 = (tid + 256) >> 4, ec1 = tid & 15;
  const _Float16* ekg0 = EKb  + (size_t)er0 * Ddim + d0 + ec0 * 4;
  const _Float16* ekg1 = EKb  + (size_t)er1 * Ddim + d0 + ec1 * 4;
  const _Float16* kvg0 = EKVb + (size_t)er0 * Ddim + d0 + ec0 * 4;
  const _Float16* kvg1 = EKVb + (size_t)er1 * Ddim + d0 + ec1 * 4;

  unsigned aoff[2], ekoff0[2], ekoff1[2], kvoff0[2], kvoff1[2];
  #pragma unroll
  for (int bb = 0; bb < 2; ++bb) {
    aoff[bb]   = (unsigned)(size_t)&sA [bb][arow * 36 + ach * 4];
    ekoff0[bb] = (unsigned)(size_t)&sEK[bb][er0 * 68 + ec0 * 4];
    ekoff1[bb] = (unsigned)(size_t)&sEK[bb][er1 * 68 + ec1 * 4];
    kvoff0[bb] = (unsigned)(size_t)&sKV[bb][er0 * 68 + ec0 * 4];
    kvoff1[bb] = (unsigned)(size_t)&sKV[bb][er1 * 68 + ec1 * 4];
  }

  const int mfrag = lane & 15;
  const int koff  = (lane >> 4) * 8;
  const int kb    = (lane >> 4) * 16;
  const int n     = wn * 16 + (lane & 15);

  v8f num = {}, den = {};

  // prologue: prefetch s-tile 0 into buffer 0
  async_b64(aoff[0],   agp);
  async_b64(ekoff0[0], ekg0);
  async_b64(ekoff1[0], ekg1);
  async_b64(kvoff0[0], kvg0);
  async_b64(kvoff1[0], kvg1);

  #pragma unroll 2
  for (int it = 0; it < Tdim / 32; ++it) {
    const int cur = it & 1;
    const int s0  = it * 32;
    if (s0 + 32 < Tdim) {
      const int nxt = cur ^ 1;
      const size_t soff = (size_t)(s0 + 32) * Ddim;
      async_b64(aoff[nxt],   agp + s0 + 32);
      async_b64(ekoff0[nxt], ekg0 + soff);
      async_b64(ekoff1[nxt], ekg1 + soff);
      async_b64(kvoff0[nxt], kvg0 + soff);
      async_b64(kvoff1[nxt], kvg1 + soff);
      wait_async5();             // only tile `it`'s group must have landed
    } else {
      wait_async0();
    }
    __syncthreads();

    v16h af, bk, bv;
    const _Float16* ar = &sA[cur][(wm * 16 + mfrag) * 36];
    #pragma unroll
    for (int j = 0; j < 8; ++j) {
      af[j]     = ar[koff + j];
      af[8 + j] = ar[16 + koff + j];
    }
    #pragma unroll
    for (int j = 0; j < 8; ++j) {
      bv[j]     = sKV[cur][(koff + j) * 68 + n];
      bk[j]     = sEK[cur][(koff + j) * 68 + n];
      bv[8 + j] = sKV[cur][(16 + koff + j) * 68 + n];
      bk[8 + j] = sEK[cur][(16 + koff + j) * 68 + n];
    }
    num = __builtin_amdgcn_wmma_f32_16x16x32_f16(false, af, false, bv,
                                                 (short)0, num, false, false);
    den = __builtin_amdgcn_wmma_f32_16x16x32_f16(false, af, false, bk,
                                                 (short)0, den, false, false);
    __syncthreads();
  }

  const int dn   = d0 + n;
  const int trow = t0 + wm * 16 + (lane >> 4) * 8;
  #pragma unroll
  for (int r = 0; r < 8; ++r) {
    size_t g = (size_t)b * TD + (size_t)(trow + r) * Ddim + dn;
    float q   = Q[g];
    float sig = 1.0f / (1.0f + __expf(-q));
    Yt[g] = (_Float16)(sig * num[r] / den[r]);
  }
}

// ---------------------------------------------------------------------------
extern "C" void kernel_launch(void* const* d_in, const int* in_sizes, int n_in,
                              void* d_out, int out_size, void* d_ws, size_t ws_size,
                              hipStream_t stream) {
  const float* x    = (const float*)d_in[0];
  const float* Wk_w = (const float*)d_in[1];
  const float* Wk_b = (const float*)d_in[2];
  const float* Wv_w = (const float*)d_in[3];
  const float* Wv_b = (const float*)d_in[4];
  const float* Wq_w = (const float*)d_in[5];
  const float* Wq_b = (const float*)d_in[6];
  const float* w    = (const float*)d_in[7];
  const float* Wo_w = (const float*)d_in[8];
  const float* Wo_b = (const float*)d_in[9];

  char* ws = (char*)d_ws;
  const size_t MB = (size_t)1 << 20;
  _Float16* xh     = (_Float16*)(ws);                    // 32 MB
  _Float16* Wkh    = (_Float16*)(ws + 32 * MB);          // 0.5 MB
  _Float16* Wvh    = (_Float16*)(ws + 32 * MB + 512 * 1024);
  _Float16* Wqh    = (_Float16*)(ws + 33 * MB);
  _Float16* Woh    = (_Float16*)(ws + 33 * MB + 512 * 1024);
  float*    Kbuf   = (float*)(ws + 34 * MB);             // 64 MB (reused as Yt)
  float*    Vbuf   = (float*)(ws + 98 * MB);             // 64 MB
  float*    Qbuf   = (float*)(ws + 162 * MB);            // 64 MB
  _Float16* expw   = (_Float16*)(ws + 226 * MB);         //  2 MB
  float*    maxK   = (float*)(ws + 228 * MB);            //  2 MB
  _Float16* EKbuf  = (_Float16*)(ws + 230 * MB);         // 32 MB
  _Float16* EKVbuf = (_Float16*)(ws + 262 * MB);         // 32 MB
  _Float16* Yth    = (_Float16*)Kbuf;                    // K dead by then

  // f32 -> f16 pre-conversion
  cvt_kernel<<<(BTD / 2) / 256, 256, 0, stream>>>(x, xh, BTD / 2);
  cvt_kernel<<<512, 256, 0, stream>>>(Wk_w, Wkh, Ddim * Ddim / 2);
  cvt_kernel<<<512, 256, 0, stream>>>(Wv_w, Wvh, Ddim * Ddim / 2);
  cvt_kernel<<<512, 256, 0, stream>>>(Wq_w, Wqh, Ddim * Ddim / 2);
  cvt_kernel<<<512, 256, 0, stream>>>(Wo_w, Woh, Ddim * Ddim / 2);

  dim3 ggrid(Bdim * Tdim / 32, Ddim / 64);               // 1024 x 8

  gemm_h<<<ggrid, 256, 0, stream>>>(xh, Wkh, Wk_b, Kbuf);
  gemm_h<<<ggrid, 256, 0, stream>>>(xh, Wvh, Wv_b, Vbuf);
  gemm_h<<<ggrid, 256, 0, stream>>>(xh, Wqh, Wq_b, Qbuf);

  maxk_kernel<<<TD / 256, 256, 0, stream>>>(Kbuf, maxK);
  expw_kernel<<<Tdim, 256, 0, stream>>>(w, expw);
  ekv_kernel<<<BTD / 256, 256, 0, stream>>>(Kbuf, Vbuf, maxK, EKbuf, EKVbuf);

  aft_attn<<<dim3(Tdim / 32, Ddim / 64, Bdim), 256, 0, stream>>>(
      expw, EKbuf, EKVbuf, Qbuf, Yth);

  gemm_h<<<ggrid, 256, 0, stream>>>(Yth, Woh, Wo_b, (float*)d_out);
}
